// MedicalAttention_87411174408330
// MI455X (gfx1250) — matrix operand order repaired and checked by reading, not compiled, not run on hardware
//
#include <hip/hip_runtime.h>
#include <hip/hip_bf16.h>

// ---------------------------------------------------------------------------
// MedicalAttention forward for gfx1250 (CDNA5, wave32, WMMA bf16 path)
// B=2, S=2048, E=768, H=12, D=64
// ---------------------------------------------------------------------------

typedef __attribute__((ext_vector_type(16))) __bf16 v16bf;
typedef __attribute__((ext_vector_type(8)))  float  v8f;

#define WMMA_BF16(a, b, c) \
  __builtin_amdgcn_wmma_f32_16x16x32_bf16(false, (a), false, (b), (short)0, (c), false, false)

// ds_swizzle_b32 group-of-32 xor pattern: offset = xor_mask<<10 | or<<5 | and(0x1f)
#define SWIZ_XOR(x, m) \
  __int_as_float(__builtin_amdgcn_ds_swizzle(__float_as_int(x), (0x1f | ((m) << 10))))

__device__ inline float xmax16(float v) {
  v = fmaxf(v, SWIZ_XOR(v, 1));
  v = fmaxf(v, SWIZ_XOR(v, 2));
  v = fmaxf(v, SWIZ_XOR(v, 4));
  v = fmaxf(v, SWIZ_XOR(v, 8));
  return v;
}
__device__ inline float xsum16(float v) {
  v += SWIZ_XOR(v, 1);
  v += SWIZ_XOR(v, 2);
  v += SWIZ_XOR(v, 4);
  v += SWIZ_XOR(v, 8);
  return v;
}

constexpr int Bn = 2, Sn = 2048, En = 768, Hn = 12, Dn = 64;
constexpr int Mtot = Bn * Sn;           // 4096 rows in all GEMMs

union V16 { uint4 u[2]; v16bf v; };

// A-matrix 16x32 bf16 layout: lane holds row M = lane&15.
// halves 0..15 -> K = base + 16*(lane>=16) + 0..15 (contiguous).
// p points at (row_ptr + kchunk + 16*hi); loads 8 bf16 at p and 8 at p+8.
__device__ inline v16bf ldA_bf(const __bf16* p) {
  V16 t;
  t.u[0] = *(const uint4*)(p);
  t.u[1] = *(const uint4*)(p + 8);
  return t.v;
}

// B-matrix 32x16 bf16 layout: lane holds col N = lane&15.
// halves 0..15 -> K = base + 16*(lane>=16) + 0..15 (contiguous).
__device__ inline v16bf ldB_bf(const __bf16* p) {
  V16 t;
  t.u[0] = *(const uint4*)(p);
  t.u[1] = *(const uint4*)(p + 8);
  return t.v;
}

// ---------------------------------------------------------------------------
// Kernel 0: bulk f32 -> bf16 conversion (8 elements / thread, b128 traffic)
// ---------------------------------------------------------------------------
__global__ void __launch_bounds__(256)
cvt_bf16_kernel(const float* __restrict__ in, __bf16* __restrict__ out, int n) {
  const int i = (blockIdx.x * 256 + threadIdx.x) * 8;
  if (i >= n) return;
  const float4 a = *(const float4*)(in + i);
  const float4 b = *(const float4*)(in + i + 4);
  union { __bf16 h[8]; uint4 u; } t;
  t.h[0] = (__bf16)a.x; t.h[1] = (__bf16)a.y; t.h[2] = (__bf16)a.z; t.h[3] = (__bf16)a.w;
  t.h[4] = (__bf16)b.x; t.h[5] = (__bf16)b.y; t.h[6] = (__bf16)b.z; t.h[7] = (__bf16)b.w;
  *(uint4*)(out + i) = t.u;
}

// ---------------------------------------------------------------------------
// Kernel 1: Y = X @ W^T + bias, bf16 in / bf16 out. 16x64 tile per wave
// (N-block == one head's D). scale folds 1/sqrt(D) into Q.
//   TRANSV==0 : store [B,H,S,D]   (Q, K)
//   TRANSV==1 : store [B,H,D,S]   (V, so attention B-operand is contiguous)
// grid = 256*12/4 = 768 blocks of 128 threads.
// ---------------------------------------------------------------------------
template <int TRANSV>
__global__ void __launch_bounds__(128)
qkv_proj_kernel(const __bf16* __restrict__ X, const __bf16* __restrict__ W,
                const float* __restrict__ bias, __bf16* __restrict__ out,
                float scale) {
  const int wave = threadIdx.x >> 5;
  const int lane = threadIdx.x & 31;
  const int hi   = lane >> 4;
  const int l15  = lane & 15;

  const int tile = blockIdx.x * 4 + wave;   // 0..3071
  const int tm   = tile / Hn;               // 16-row block
  const int hb   = tile % Hn;               // head == 64-wide N block

  const __bf16* arow = X + (size_t)(tm * 16 + l15) * En;
  const __bf16* b0 = W + (size_t)(hb * 64 + 0 * 16 + l15) * En;
  const __bf16* b1 = W + (size_t)(hb * 64 + 1 * 16 + l15) * En;
  const __bf16* b2 = W + (size_t)(hb * 64 + 2 * 16 + l15) * En;
  const __bf16* b3 = W + (size_t)(hb * 64 + 3 * 16 + l15) * En;

  v8f c0 = {0.f,0.f,0.f,0.f,0.f,0.f,0.f,0.f};
  v8f c1 = c0, c2 = c0, c3 = c0;
  for (int k = 0; k < En; k += 32) {
    const v16bf a = ldA_bf(arow + k + 16 * hi);
    c0 = WMMA_BF16(a, ldB_bf(b0 + k + 16 * hi), c0);
    c1 = WMMA_BF16(a, ldB_bf(b1 + k + 16 * hi), c1);
    c2 = WMMA_BF16(a, ldB_bf(b2 + k + 16 * hi), c2);
    c3 = WMMA_BF16(a, ldB_bf(b3 + k + 16 * hi), c3);
  }

  const float bi0 = bias[hb * 64 + 0 * 16 + l15];
  const float bi1 = bias[hb * 64 + 1 * 16 + l15];
  const float bi2 = bias[hb * 64 + 2 * 16 + l15];
  const float bi3 = bias[hb * 64 + 3 * 16 + l15];

#pragma unroll
  for (int i = 0; i < 8; ++i) {
    const int m  = tm * 16 + i + 8 * hi;    // row in [0, 4096)
    const int bb = m / Sn;
    const int ss = m % Sn;
    const size_t base = (size_t)(bb * Hn + hb) * (size_t)Sn * Dn;
    if (TRANSV) {
      out[base + (size_t)(0 * 16 + l15) * Sn + ss] = (__bf16)((c0[i] + bi0) * scale);
      out[base + (size_t)(1 * 16 + l15) * Sn + ss] = (__bf16)((c1[i] + bi1) * scale);
      out[base + (size_t)(2 * 16 + l15) * Sn + ss] = (__bf16)((c2[i] + bi2) * scale);
      out[base + (size_t)(3 * 16 + l15) * Sn + ss] = (__bf16)((c3[i] + bi3) * scale);
    } else {
      __bf16* op = out + base + (size_t)ss * Dn;
      op[0 * 16 + l15] = (__bf16)((c0[i] + bi0) * scale);
      op[1 * 16 + l15] = (__bf16)((c1[i] + bi1) * scale);
      op[2 * 16 + l15] = (__bf16)((c2[i] + bi2) * scale);
      op[3 * 16 + l15] = (__bf16)((c3[i] + bi3) * scale);
    }
  }
}

// ---------------------------------------------------------------------------
// Kernel 2: flash attention per (b,h). One wave owns 16 query rows.
// Q pre-scaled by 1/8. Per-key additive bias mcw[h]*mask[b,key].
// Row max reduced per chunk with ds_swizzle; denominator kept lane-local and
// reduced once at the end (alpha is lane-uniform after the max reduction).
// grid = (S/16/4, B*H), block = 128 (4 waves).
// ---------------------------------------------------------------------------
__global__ void __launch_bounds__(128)
flash_attn_kernel(const __bf16* __restrict__ Q, const __bf16* __restrict__ K,
                  const __bf16* __restrict__ Vt, const float* __restrict__ mask,
                  const float* __restrict__ mcw, __bf16* __restrict__ out) {
  __shared__ __align__(32) __bf16 lds[4][16 * 32];   // P staging, per wave

  const int wave = threadIdx.x >> 5;
  const int lane = threadIdx.x & 31;
  const int hi   = lane >> 4;
  const int l15  = lane & 15;

  const int bh = blockIdx.y;
  const int b  = bh / Hn;
  const int h  = bh % Hn;
  const int qbase = (blockIdx.x * 4 + wave) * 16;

  const __bf16* Qp = Q  + (size_t)bh * Sn * Dn;
  const __bf16* Kp = K  + (size_t)bh * Sn * Dn;
  const __bf16* Vp = Vt + (size_t)bh * Dn * Sn;
  const float w = mcw[h];

  // Load the 16x64 Q tile once (A layout, two K=32 chunks).
  const __bf16* qrow = Qp + (size_t)(qbase + l15) * Dn;
  const v16bf aq0 = ldA_bf(qrow + 0  + 16 * hi);
  const v16bf aq1 = ldA_bf(qrow + 32 + 16 * hi);

  v8f o0 = {0.f,0.f,0.f,0.f,0.f,0.f,0.f,0.f};
  v8f o1 = o0, o2 = o0, o3 = o0;
  float m_s[8], l_loc[8];                  // l_loc: lane-local partial denom
#pragma unroll
  for (int i = 0; i < 8; ++i) { m_s[i] = -3.0e38f; l_loc[i] = 0.f; }

  __bf16* lp = lds[wave];

  for (int kc = 0; kc < Sn; kc += 32) {
    // ---- scores: two 16-key tiles, K-dim 64 = 2 WMMAs each ----
    v8f s0 = {0.f,0.f,0.f,0.f,0.f,0.f,0.f,0.f};
    v8f s1 = s0;
    const __bf16* krow0 = Kp + (size_t)(kc + l15) * Dn;
    const __bf16* krow1 = Kp + (size_t)(kc + 16 + l15) * Dn;
    s0 = WMMA_BF16(aq0, ldB_bf(krow0 + 0  + 16 * hi), s0);
    s0 = WMMA_BF16(aq1, ldB_bf(krow0 + 32 + 16 * hi), s0);
    s1 = WMMA_BF16(aq0, ldB_bf(krow1 + 0  + 16 * hi), s1);
    s1 = WMMA_BF16(aq1, ldB_bf(krow1 + 32 + 16 * hi), s1);

    // Prefetch next chunk's K and V lines (speculative; OOB tail is dropped).
    __builtin_prefetch(krow0 + 32 * Dn, 0, 1);
    __builtin_prefetch(krow1 + 32 * Dn, 0, 1);
    __builtin_prefetch(Vp + (size_t)l15 * Sn + kc + 32, 0, 1);

    const float bias0 = w * mask[(size_t)b * Sn + kc + l15];
    const float bias1 = w * mask[(size_t)b * Sn + kc + 16 + l15];

    // ---- online softmax; only the row max needs cross-lane traffic ----
#pragma unroll
    for (int i = 0; i < 8; ++i) {
      const float a0 = s0[i] + bias0;
      const float a1 = s1[i] + bias1;
      const float mx = xmax16(fmaxf(a0, a1));
      const float mnew  = fmaxf(m_s[i], mx);
      const float alpha = __expf(m_s[i] - mnew);   // lane-uniform
      m_s[i] = mnew;
      const float p0 = __expf(a0 - mnew);
      const float p1 = __expf(a1 - mnew);
      s0[i] = p0; s1[i] = p1;
      l_loc[i] = l_loc[i] * alpha + (p0 + p1);
      o0[i] *= alpha; o1[i] *= alpha; o2[i] *= alpha; o3[i] *= alpha;
    }

    // ---- transpose P (C layout) -> A layout via LDS ----
#pragma unroll
    for (int i = 0; i < 8; ++i) {
      const int row = i + 8 * hi;
      lp[row * 32 + l15]      = (__bf16)s0[i];
      lp[row * 32 + 16 + l15] = (__bf16)s1[i];
    }
    __syncthreads();
    const v16bf ap = ldA_bf(lp + l15 * 32 + 16 * hi);
    __syncthreads();

    // ---- O += P @ V : 4 N-tiles of 16, V transposed so keys contiguous ----
    o0 = WMMA_BF16(ap, ldB_bf(Vp + (size_t)(0 * 16 + l15) * Sn + kc + 16 * hi), o0);
    o1 = WMMA_BF16(ap, ldB_bf(Vp + (size_t)(1 * 16 + l15) * Sn + kc + 16 * hi), o1);
    o2 = WMMA_BF16(ap, ldB_bf(Vp + (size_t)(2 * 16 + l15) * Sn + kc + 16 * hi), o2);
    o3 = WMMA_BF16(ap, ldB_bf(Vp + (size_t)(3 * 16 + l15) * Sn + kc + 16 * hi), o3);
  }

  // ---- final denominator reduction, normalize, store bf16 [B, S, E] ----
#pragma unroll
  for (int i = 0; i < 8; ++i) {
    const float inv = 1.0f / xsum16(l_loc[i]);
    const int row = qbase + i + 8 * hi;
    __bf16* op = out + ((size_t)b * Sn + row) * En + h * Dn;
    op[0 * 16 + l15] = (__bf16)(o0[i] * inv);
    op[1 * 16 + l15] = (__bf16)(o1[i] * inv);
    op[2 * 16 + l15] = (__bf16)(o2[i] * inv);
    op[3 * 16 + l15] = (__bf16)(o3[i] * inv);
  }
}

// ---------------------------------------------------------------------------
// Kernel 3: out = attn(bf16) @ Wo^T + bo, f32 result. 16x64 tile per wave.
// ---------------------------------------------------------------------------
__global__ void __launch_bounds__(128)
out_proj_kernel(const __bf16* __restrict__ A, const __bf16* __restrict__ W,
                const float* __restrict__ bias, float* __restrict__ out) {
  const int wave = threadIdx.x >> 5;
  const int lane = threadIdx.x & 31;
  const int hi   = lane >> 4;
  const int l15  = lane & 15;

  const int tile = blockIdx.x * 4 + wave;   // 0..3071
  const int tm   = tile / Hn;
  const int nb   = tile % Hn;               // 64-wide N block

  const __bf16* arow = A + (size_t)(tm * 16 + l15) * En;
  const __bf16* b0 = W + (size_t)(nb * 64 + 0 * 16 + l15) * En;
  const __bf16* b1 = W + (size_t)(nb * 64 + 1 * 16 + l15) * En;
  const __bf16* b2 = W + (size_t)(nb * 64 + 2 * 16 + l15) * En;
  const __bf16* b3 = W + (size_t)(nb * 64 + 3 * 16 + l15) * En;

  v8f c0 = {0.f,0.f,0.f,0.f,0.f,0.f,0.f,0.f};
  v8f c1 = c0, c2 = c0, c3 = c0;
  for (int k = 0; k < En; k += 32) {
    const v16bf a = ldA_bf(arow + k + 16 * hi);
    c0 = WMMA_BF16(a, ldB_bf(b0 + k + 16 * hi), c0);
    c1 = WMMA_BF16(a, ldB_bf(b1 + k + 16 * hi), c1);
    c2 = WMMA_BF16(a, ldB_bf(b2 + k + 16 * hi), c2);
    c3 = WMMA_BF16(a, ldB_bf(b3 + k + 16 * hi), c3);
  }

  const float bi0 = bias[nb * 64 + 0 * 16 + l15];
  const float bi1 = bias[nb * 64 + 1 * 16 + l15];
  const float bi2 = bias[nb * 64 + 2 * 16 + l15];
  const float bi3 = bias[nb * 64 + 3 * 16 + l15];

#pragma unroll
  for (int i = 0; i < 8; ++i) {
    const int m = tm * 16 + i + 8 * hi;
    float* op = out + (size_t)m * En + nb * 64;
    op[0 * 16 + l15] = c0[i] + bi0;
    op[1 * 16 + l15] = c1[i] + bi1;
    op[2 * 16 + l15] = c2[i] + bi2;
    op[3 * 16 + l15] = c3[i] + bi3;
  }
}

// ---------------------------------------------------------------------------
// Launch
// ---------------------------------------------------------------------------
extern "C" void kernel_launch(void* const* d_in, const int* in_sizes, int n_in,
                              void* d_out, int out_size, void* d_ws, size_t ws_size,
                              hipStream_t stream) {
  const float* query = (const float*)d_in[0];
  const float* key_i = (const float*)d_in[1];
  const float* value = (const float*)d_in[2];
  const float* mask  = (const float*)d_in[3];
  const float* Wq    = (const float*)d_in[4];
  const float* bq    = (const float*)d_in[5];
  const float* Wk    = (const float*)d_in[6];
  const float* bk    = (const float*)d_in[7];
  const float* Wv    = (const float*)d_in[8];
  const float* bv    = (const float*)d_in[9];
  const float* Wo    = (const float*)d_in[10];
  const float* bo    = (const float*)d_in[11];
  const float* mcw   = (const float*)d_in[12];

  // Workspace layout (all bf16):
  //   Q, K [B,H,S,D], Vt [B,H,D,S], attn-out [B,S,E]   : 4 x 3,145,728
  //   xq, xk, xv (converted activations [B,S,E])       : 3 x 3,145,728
  //   wqb, wkb, wvb, wob (converted weights [E,E])     : 4 x   589,824
  const size_t act = (size_t)Mtot * En;   // 3,145,728
  const size_t wel = (size_t)En * En;     //   589,824
  __bf16* qws = (__bf16*)d_ws;
  __bf16* kws = qws + act;
  __bf16* vws = kws + act;
  __bf16* aws = vws + act;
  __bf16* xq  = aws + act;
  __bf16* xk  = xq  + act;
  __bf16* xv  = xk  + act;
  __bf16* wqb = xv  + act;
  __bf16* wkb = wqb + wel;
  __bf16* wvb = wkb + wel;
  __bf16* wob = wvb + wel;

  const dim3 cblk(256);
  const dim3 gAct((unsigned)(act / 8 / 256));   // 1536
  const dim3 gWgt((unsigned)(wel / 8 / 256));   //  288
  cvt_bf16_kernel<<<gAct, cblk, 0, stream>>>(query, xq, (int)act);
  cvt_bf16_kernel<<<gAct, cblk, 0, stream>>>(key_i, xk, (int)act);
  cvt_bf16_kernel<<<gAct, cblk, 0, stream>>>(value, xv, (int)act);
  cvt_bf16_kernel<<<gWgt, cblk, 0, stream>>>(Wq, wqb, (int)wel);
  cvt_bf16_kernel<<<gWgt, cblk, 0, stream>>>(Wk, wkb, (int)wel);
  cvt_bf16_kernel<<<gWgt, cblk, 0, stream>>>(Wv, wvb, (int)wel);
  cvt_bf16_kernel<<<gWgt, cblk, 0, stream>>>(Wo, wob, (int)wel);

  const dim3 blk(128);
  const dim3 gProj((Mtot / 16) * Hn / 4);       // 768 blocks, 16x64 tile/wave
  const float qscale = 0.125f;                  // 1/sqrt(64)

  qkv_proj_kernel<0><<<gProj, blk, 0, stream>>>(xq, wqb, bq, qws, qscale);
  qkv_proj_kernel<0><<<gProj, blk, 0, stream>>>(xk, wkb, bk, kws, 1.0f);
  qkv_proj_kernel<1><<<gProj, blk, 0, stream>>>(xv, wvb, bv, vws, 1.0f);

  const dim3 gAttn(Sn / 16 / 4, Bn * Hn);       // (32, 24)
  flash_attn_kernel<<<gAttn, blk, 0, stream>>>(qws, kws, vws, mask, mcw, aws);

  out_proj_kernel<<<gProj, blk, 0, stream>>>(aws, wob, bo, (float*)d_out);
}
